// Attention_78546361909795
// MI455X (gfx1250) — matrix operand order repaired and checked by reading, not compiled, not run on hardware
//
#include <hip/hip_runtime.h>
#include <hip/hip_bf16.h>
#include <math.h>

// ---------------------------------------------------------------------------
// Attention: y = OutProj( Softmax(Q K^T * scale) V ), QKV = x @ W_qkv^T
// B=8, S=2048, D=512, H=8, DH=64. All matmuls on v_wmma_f32_16x16x32_f16.
// Shared operand tiles staged through LDS with async-to-LDS copies
// (double buffered), per-wave A-fragments register double-buffered.
// ---------------------------------------------------------------------------

typedef __attribute__((ext_vector_type(16))) _Float16 v16h;
typedef __attribute__((ext_vector_type(8)))  _Float16 v8h;
typedef __attribute__((ext_vector_type(4)))  _Float16 v4h;
typedef __attribute__((ext_vector_type(8)))  float    v8f;
typedef __attribute__((ext_vector_type(4)))  int      v4i;

#define NB   8
#define NS   2048
#define ND   512
#define NH   8
#define NDH  64
#define NE   1536          // 3 * H * DH
#define NM   (NB * NS)     // 16384 rows
#define SCALE_LOG2E 0.1803368801111204f   // 64^-0.5 * log2(e)

#if defined(__has_builtin)
#  if __has_builtin(__builtin_amdgcn_global_load_async_to_lds_b128)
#    define HAVE_ASYNC_LDS 1
#  endif
#endif
#ifndef HAVE_ASYNC_LDS
#  define HAVE_ASYNC_LDS 0
#endif

// pointer-to-int4 in explicit address spaces (builtin's expected types)
typedef __attribute__((address_space(1))) v4i* as1_v4i_p;
typedef __attribute__((address_space(3))) v4i* as3_v4i_p;

// ---- async 16-byte global->LDS copy (per lane) ----------------------------
__device__ __forceinline__ void async_copy16(const _Float16* g, _Float16* l) {
#if HAVE_ASYNC_LDS
  __builtin_amdgcn_global_load_async_to_lds_b128(
      (as1_v4i_p)(unsigned long long)(const void*)g,
      (as3_v4i_p)(unsigned int)(unsigned long long)(void*)l,
      0, 0);
#else
  *(v8h*)l = *(const v8h*)g;   // fallback: sync load + ds_store
#endif
}

__device__ __forceinline__ void wait_async_done() {
#if HAVE_ASYNC_LDS
#  if __has_builtin(__builtin_amdgcn_s_wait_asynccnt)
  __builtin_amdgcn_s_wait_asynccnt(0);
#  else
  asm volatile("s_wait_asynccnt 0x0" ::: "memory");
#  endif
#endif
}

// stage a 64x32 f16 tile (row-major, leading dim ld) into LDS[64][32];
// 256 16B chunks spread over the 128 block threads
__device__ __forceinline__ void stage_tile_64x32(const _Float16* __restrict__ g,
                                                 int ld, _Float16* l, int tid) {
#pragma unroll
  for (int c = 0; c < 2; ++c) {
    const int chunk = tid + c * 128;      // 0..255
    const int r  = chunk >> 2;            // 4 chunks per 32-elem row
    const int co = (chunk & 3) * 8;
    async_copy16(g + r * ld + co, l + r * 32 + co);
  }
}

// stage a 32x64 f16 tile into LDS[32][64]
__device__ __forceinline__ void stage_tile_32x64(const _Float16* __restrict__ g,
                                                 int ld, _Float16* l, int tid) {
#pragma unroll
  for (int c = 0; c < 2; ++c) {
    const int chunk = tid + c * 128;
    const int r  = chunk >> 3;            // 8 chunks per 64-elem row
    const int co = (chunk & 7) * 8;
    async_copy16(g + r * ld + co, l + r * 64 + co);
  }
}

// ---- WMMA fragment loaders (row-major f16 source; works for global & LDS) -
// A-matrix 16x32: lane l -> row l%16; K = half*16 .. +15 (contiguous).
__device__ __forceinline__ v16h load_frag_a(const _Float16* base, int ld, int lane) {
  const int half = lane >> 4, r = lane & 15;
  const _Float16* p = base + r * ld + half * 16;
  union { v16h v; v8h h[2]; } u;
  u.h[0] = *(const v8h*)(p);
  u.h[1] = *(const v8h*)(p + 8);
  return u.v;
}

// B-matrix 32x16: lane l -> column l%16 == row (l%16) of transposed operand;
// K = half*16 .. +15 (contiguous 32B).
__device__ __forceinline__ v16h load_frag_b(const _Float16* base, int ld, int lane) {
  const int half = lane >> 4, r = lane & 15;
  return *(const v16h*)(base + r * ld + half * 16);
}

#define WMMA_F16(A, B, C) \
  __builtin_amdgcn_wmma_f32_16x16x32_f16(false, (A), false, (B), (short)0, (C), false, false)

// ---- f32 -> f16 conversion (4 elems/thread) -------------------------------
__global__ void cvt_f32_f16_kernel(const float* __restrict__ src,
                                   _Float16* __restrict__ dst, int n) {
  const int i = (blockIdx.x * blockDim.x + threadIdx.x) * 4;
  if (i < n) {
    const float4 f = *(const float4*)(src + i);
    v4h h; h[0] = (_Float16)f.x; h[1] = (_Float16)f.y;
    h[2] = (_Float16)f.z; h[3] = (_Float16)f.w;
    *(v4h*)(dst + i) = h;
  }
}

// ---- shared 32(M) x 64(N) per-wave GEMM mainloop, acc = A @ W^T -----------
// W tile (64 rows x 32 k) staged in LDS (double buffered, shared by 4 waves);
// A fragments register double-buffered.
__device__ __forceinline__ void gemm32x64_mainloop(const _Float16* __restrict__ a0p,
                                                   const _Float16* __restrict__ a1p,
                                                   const _Float16* __restrict__ wp,
                                                   _Float16* sw, int tid,
                                                   v8f acc[8]) {
  const int lane = tid & 31;
  stage_tile_64x32(wp, ND, sw, tid);            // stage k0 = 0
  v16h a0 = load_frag_a(a0p, ND, lane);
  v16h a1 = load_frag_a(a1p, ND, lane);
  for (int it = 0; it < ND / 32; ++it) {
    const int k0 = it * 32;
    wait_async_done();
    __syncthreads();
    const _Float16* wl = sw + (it & 1) * (64 * 32);
    if (k0 + 32 < ND)
      stage_tile_64x32(wp + k0 + 32, ND, sw + ((it + 1) & 1) * (64 * 32), tid);
    const int kn = (k0 + 32 < ND) ? (k0 + 32) : 0;   // harmless refetch on last
    const v16h na0 = load_frag_a(a0p + kn, ND, lane);
    const v16h na1 = load_frag_a(a1p + kn, ND, lane);
#pragma unroll
    for (int t = 0; t < 4; ++t) {
      const v16h b = load_frag_b(wl + t * 16 * 32, 32, lane);
      acc[t]     = WMMA_F16(a0, b, acc[t]);
      acc[4 + t] = WMMA_F16(a1, b, acc[4 + t]);
    }
    a0 = na0; a1 = na1;
  }
}

// ---- QKV projection: qkv = x @ W_qkv^T scattered into Q,K (B,H,S,DH) and
// V^T (B,H,DH,S). Block = 4 waves = 128(M) x 64(N). ------------------------
__global__ __launch_bounds__(128)
void qkv_gemm_kernel(const _Float16* __restrict__ x,
                     const _Float16* __restrict__ w,
                     _Float16* __restrict__ qbuf,
                     _Float16* __restrict__ kbuf,
                     _Float16* __restrict__ vtbuf) {
  __shared__ _Float16 sw[2 * 64 * 32];
  const int tid = threadIdx.x, lane = tid & 31, wave = tid >> 5;
  const int m0 = blockIdx.x * 128 + wave * 32;
  const int n0 = blockIdx.y * 64;

  v8f acc[8] = {};
  gemm32x64_mainloop(x + (size_t)m0 * ND, x + (size_t)(m0 + 16) * ND,
                     w + (size_t)n0 * ND, sw, tid, acc);

  const int half = lane >> 4, col = lane & 15;
#pragma unroll
  for (int mt = 0; mt < 2; ++mt)
#pragma unroll
    for (int t = 0; t < 4; ++t) {
      const int n  = n0 + t * 16 + col;
      const int h  = n / (3 * NDH);
      const int tt = n % (3 * NDH);
      const int which = tt / NDH, d = tt % NDH;
#pragma unroll
      for (int i = 0; i < 8; ++i) {
        const int m = m0 + mt * 16 + half * 8 + i;
        const int b = m >> 11, s = m & (NS - 1);
        const _Float16 val = (_Float16)acc[mt * 4 + t][i];
        if (which == 0)
          qbuf[(((b * NH + h) * NS) + s) * NDH + d] = val;
        else if (which == 1)
          kbuf[(((b * NH + h) * NS) + s) * NDH + d] = val;
        else
          vtbuf[(((b * NH + h) * NDH) + d) * NS + s] = val;
      }
    }
}

// ---- Flash attention: block = 4 waves x 16 queries of one (b,h). ----------
// K (32x64) and V^T (64x32) tiles async-staged to LDS, double buffered and
// shared by all 4 waves. scores^T = K_tile x Q^T so each lane's accumulators
// hold one query -> softmax = in-lane reduce + one xor-16 shuffle.
__global__ __launch_bounds__(128)
void attn_kernel(const _Float16* __restrict__ qg,
                 const _Float16* __restrict__ kg,
                 const _Float16* __restrict__ vtg,
                 _Float16* __restrict__ attO) {
  __shared__ _Float16 sK[2 * 32 * 64];
  __shared__ _Float16 sV[2 * 64 * 32];
  const int tid = threadIdx.x, lane = tid & 31, wave = tid >> 5;
  const int bh = blockIdx.y;
  const int s0 = (blockIdx.x * 4 + wave) * 16;
  const int b = bh >> 3, h = bh & 7;

  const _Float16* Q  = qg  + ((size_t)bh * NS + s0) * NDH;
  const _Float16* K  = kg  + (size_t)bh * NS * NDH;
  const _Float16* VT = vtg + (size_t)bh * NDH * NS;

  const v16h qb0 = load_frag_b(Q + 0,  NDH, lane);   // Q^T, d[0..31]
  const v16h qb1 = load_frag_b(Q + 32, NDH, lane);   // Q^T, d[32..63]

  v8f   o[4] = {};                  // O^T tiles: rows d (4x16), cols q (16)
  float mrun = -3.0e38f, lrun = 0.f;

  stage_tile_32x64(K,  NDH, sK, tid);      // keys 0..31
  stage_tile_64x32(VT, NS,  sV, tid);      // V^T cols 0..31

  for (int it = 0; it < NS / 32; ++it) {
    const int kb = it * 32;
    wait_async_done();
    __syncthreads();
    const _Float16* Kl = sK + (it & 1) * (32 * 64);
    const _Float16* Vl = sV + (it & 1) * (64 * 32);
    if (kb + 32 < NS) {
      stage_tile_32x64(K + (kb + 32) * NDH, NDH,
                       sK + ((it + 1) & 1) * (32 * 64), tid);
      stage_tile_64x32(VT + (kb + 32), NS,
                       sV + ((it + 1) & 1) * (64 * 32), tid);
    }

    // scores^T tiles from LDS: tile0 keys kb..kb+15, tile1 kb+16..kb+31
    v16h ka0 = load_frag_a(Kl + 0,  64, lane);
    v16h ka1 = load_frag_a(Kl + 32, 64, lane);
    v8f st0 = WMMA_F16(ka0, qb0, (v8f){});
    st0 = WMMA_F16(ka1, qb1, st0);
    ka0 = load_frag_a(Kl + 16 * 64 + 0,  64, lane);
    ka1 = load_frag_a(Kl + 16 * 64 + 32, 64, lane);
    v8f st1 = WMMA_F16(ka0, qb0, (v8f){});
    st1 = WMMA_F16(ka1, qb1, st1);

    // ---- base-2 online softmax (per lane: 16 keys of one query) ----
    float p[16];
    float mblk = -3.0e38f;
#pragma unroll
    for (int i = 0; i < 8; ++i) {
      p[i]     = st0[i] * SCALE_LOG2E;
      p[8 + i] = st1[i] * SCALE_LOG2E;
      mblk = fmaxf(mblk, fmaxf(p[i], p[8 + i]));
    }
    mblk = fmaxf(mblk, __shfl_xor(mblk, 16, 32));
    const float mnew = fmaxf(mrun, mblk);
    const float c = exp2f(mrun - mnew);
    float sum = 0.f;
#pragma unroll
    for (int i = 0; i < 16; ++i) { p[i] = exp2f(p[i] - mnew); sum += p[i]; }
    sum += __shfl_xor(sum, 16, 32);
    lrun = lrun * c + sum;
    mrun = mnew;
#pragma unroll
    for (int dt = 0; dt < 4; ++dt)
#pragma unroll
      for (int i = 0; i < 8; ++i) o[dt][i] *= c;

    // ---- rebuild P^T as a 32x16 B-fragment ----
    // lanes<16: k[0..15] = own tile0 (0..7) + partner tile0 (8..15)
    // lanes>=16: k[16..31] = partner tile1 (16..23) + own tile1 (24..31)
    union U8 { v8h v; _Float16 e[8]; int i4[4]; };
    U8 lo, hi, loX, hiX;
#pragma unroll
    for (int j = 0; j < 8; ++j) {
      lo.e[j] = (_Float16)p[j];
      hi.e[j] = (_Float16)p[8 + j];
    }
#pragma unroll
    for (int j = 0; j < 4; ++j) {
      loX.i4[j] = __shfl_xor(lo.i4[j], 16, 32);
      hiX.i4[j] = __shfl_xor(hi.i4[j], 16, 32);
    }
    union { v16h v; v8h hh[2]; } pf;
    if (lane < 16) { pf.hh[0] = lo.v;  pf.hh[1] = loX.v; }
    else           { pf.hh[0] = hiX.v; pf.hh[1] = hi.v;  }

    // ---- O^T += V^T x P^T (from LDS) ----
#pragma unroll
    for (int dt = 0; dt < 4; ++dt) {
      const v16h va = load_frag_a(Vl + (dt * 16) * 32, 32, lane);
      o[dt] = WMMA_F16(va, pf.v, o[dt]);
    }
  }

  // normalize; store to (B,S,H*DH) f16 for the output projection
  const float inv = 1.0f / lrun;
  const int half = lane >> 4, col = lane & 15;
  const int s = s0 + col;
#pragma unroll
  for (int dt = 0; dt < 4; ++dt)
#pragma unroll
    for (int i = 0; i < 8; ++i) {
      const int d = dt * 16 + half * 8 + i;
      attO[((size_t)(b * NS + s)) * (NH * NDH) + h * NDH + d] =
          (_Float16)(o[dt][i] * inv);
    }
}

// ---- Output projection: y = attO @ W_out^T + b_out (f32 out) --------------
__global__ __launch_bounds__(128)
void out_gemm_kernel(const _Float16* __restrict__ a,
                     const _Float16* __restrict__ w,
                     const float* __restrict__ bias,
                     float* __restrict__ y) {
  __shared__ _Float16 sw[2 * 64 * 32];
  const int tid = threadIdx.x, lane = tid & 31, wave = tid >> 5;
  const int m0 = blockIdx.x * 128 + wave * 32;
  const int n0 = blockIdx.y * 64;

  v8f acc[8] = {};
  gemm32x64_mainloop(a + (size_t)m0 * ND, a + (size_t)(m0 + 16) * ND,
                     w + (size_t)n0 * ND, sw, tid, acc);

  const int half = lane >> 4, col = lane & 15;
#pragma unroll
  for (int mt = 0; mt < 2; ++mt)
#pragma unroll
    for (int t = 0; t < 4; ++t) {
      const int n = n0 + t * 16 + col;
      const float bv = bias[n];
#pragma unroll
      for (int i = 0; i < 8; ++i) {
        const int m = m0 + mt * 16 + half * 8 + i;
        y[(size_t)m * ND + n] = acc[mt * 4 + t][i] + bv;
      }
    }
}

// ---------------------------------------------------------------------------
extern "C" void kernel_launch(void* const* d_in, const int* in_sizes, int n_in,
                              void* d_out, int out_size, void* d_ws, size_t ws_size,
                              hipStream_t stream) {
  (void)in_sizes; (void)n_in; (void)out_size; (void)ws_size;
  const float* x    = (const float*)d_in[0];   // (B,S,D)
  const float* Wqkv = (const float*)d_in[1];   // (3*INNER, D)
  const float* Wout = (const float*)d_in[2];   // (D, INNER)
  const float* bout = (const float*)d_in[3];   // (D,)
  float* y = (float*)d_out;                    // (B,S,D)

  // workspace carve-up (all f16)
  _Float16* xh    = (_Float16*)d_ws;                 // NM*ND
  _Float16* wqkvh = xh    + (size_t)NM * ND;         // NE*ND
  _Float16* wouth = wqkvh + (size_t)NE * ND;         // ND*ND
  _Float16* qb    = wouth + (size_t)ND * ND;         // B*H*S*DH
  _Float16* kb    = qb    + (size_t)NB * NH * NS * NDH;
  _Float16* vtb   = kb    + (size_t)NB * NH * NS * NDH;
  _Float16* aO    = vtb   + (size_t)NB * NH * NS * NDH;  // NM * (H*DH)

  const int nx = NM * ND, nwq = NE * ND, nwo = ND * ND;
  cvt_f32_f16_kernel<<<(nx / 4 + 255) / 256, 256, 0, stream>>>(x,    xh,    nx);
  cvt_f32_f16_kernel<<<(nwq / 4 + 255) / 256, 256, 0, stream>>>(Wqkv, wqkvh, nwq);
  cvt_f32_f16_kernel<<<(nwo / 4 + 255) / 256, 256, 0, stream>>>(Wout, wouth, nwo);

  qkv_gemm_kernel<<<dim3(NM / 128, NE / 64), 128, 0, stream>>>(xh, wqkvh, qb, kb, vtb);
  attn_kernel<<<dim3(NS / 64, NB * NH), 128, 0, stream>>>(qb, kb, vtb, aO);
  out_gemm_kernel<<<dim3(NM / 128, ND / 64), 128, 0, stream>>>(aO, wouth, bout, y);
}
